// SSMBlockFast_231928234634
// MI455X (gfx1250) — compile-verified, matrix operand order checked by reading, not run on hardware
//
#include <hip/hip_runtime.h>
#include <stdint.h>

typedef __attribute__((ext_vector_type(16))) __bf16 v16bf;
typedef __attribute__((ext_vector_type(8)))  float  v8f;

#define DM 512
#define DS 16

// ---------- fp32 -> bf16 (round to nearest even) ----------
__device__ __forceinline__ uint16_t f2bf(float f) {
    union { float f; uint32_t u; } c; c.f = f;
    uint32_t u = c.u;
    uint32_t r = u + 0x7FFFu + ((u >> 16) & 1u);
    return (uint16_t)(r >> 16);
}

__global__ void k_cvt(const float* __restrict__ src, uint16_t* __restrict__ dst, int n) {
    int i = blockIdx.x * blockDim.x + threadIdx.x;
    if (i < n) dst[i] = f2bf(src[i]);
}

// ---------- discretization constants ----------
__global__ void k_setup(const float* __restrict__ A_log,
                        float* __restrict__ Abar, float* __restrict__ Bbar, float dt) {
    int s = threadIdx.x;
    if (s < DS) {
        float A  = -fminf(fmaxf(__expf(A_log[s]), 1e-8f), 10.0f);
        float ab = __expf(dt * A);
        Abar[s] = ab;
        Bbar[s] = (fabsf(A) > 1e-8f) ? (ab - 1.0f) / (A + 1e-8f) : dt;
    }
}

// Fragment layout per CDNA5 ISA 7.12.2 (16-bit A 16x32):
// lane L: row/col = L%16, half = L/16; element pair g (=j>>1):
//   K = (g&3)*2 + (g>>2)*16 + half*8  (pair K,K+1 contiguous -> dword load)
union Frag { v16bf v; uint32_t u[8]; };

__device__ __forceinline__ v8f wmma_bf16(const Frag& a, const Frag& b, v8f c) {
    return __builtin_amdgcn_wmma_f32_16x16x32_bf16(
        false, a.v, false, b.v, (short)0, c, false, false);
}

// ---------- gate GEMM + bias + sigmoid epilogue ----------
// C[M x ncols] = sigmoid(A[M x K] * B^T + bias), B stored [ncols x K].
// 8 waves/block, each wave owns a 32x32 output (2x2 WMMA tiles): A frags
// reused across 2 N-tiles, B frags across 2 M-tiles -> 8 B128 loads / 4 WMMA.
__global__ void k_gemm_gate(const uint16_t* __restrict__ Abf,
                            const uint16_t* __restrict__ Bbf,
                            const float* __restrict__ bias,
                            float* __restrict__ G, int K, int ncols) {
    int lane = threadIdx.x & 31;
    int w    = threadIdx.x >> 5;
    int wm   = w & 1, wn = w >> 1;           // 2 x 4 wave grid -> 64 x 128 macro-tile
    int m0   = blockIdx.x * 64  + wm * 32;
    int n0   = blockIdx.y * 128 + wn * 32;
    int r    = lane & 15;
    int half = lane >> 4;

    const uint16_t* a0 = Abf + (size_t)(m0 + r) * K;
    const uint16_t* a1 = a0 + (size_t)16 * K;
    const uint16_t* b0 = Bbf + (size_t)(n0 + r) * K;
    const uint16_t* b1 = b0 + (size_t)16 * K;

    v8f c00 = {}, c01 = {}, c10 = {}, c11 = {};

    for (int k0 = 0; k0 < K; k0 += 32) {
        Frag fa0, fa1, fb0, fb1;
#pragma unroll
        for (int g = 0; g < 8; ++g) {
            int kk = k0 + ((g & 3) * 2) + ((g >> 2) * 16) + half * 8;
            fa0.u[g] = *(const uint32_t*)(a0 + kk);
            fa1.u[g] = *(const uint32_t*)(a1 + kk);
            fb0.u[g] = *(const uint32_t*)(b0 + kk);
            fb1.u[g] = *(const uint32_t*)(b1 + kk);
        }
        c00 = wmma_bf16(fa0, fb0, c00);
        c01 = wmma_bf16(fa0, fb1, c01);
        c10 = wmma_bf16(fa1, fb0, c10);
        c11 = wmma_bf16(fa1, fb1, c11);
    }

    // epilogue: gate = sigmoid(logit + bias)
    v8f accs[4] = { c00, c01, c10, c11 };
#pragma unroll
    for (int t = 0; t < 4; ++t) {
        int mbase = m0 + (t >> 1) * 16;
        int nn    = n0 + (t & 1) * 16 + r;
        float bv  = bias[nn];
#pragma unroll
        for (int i = 0; i < 8; ++i) {
            int mm = mbase + i + half * 8;
            float lg = accs[t][i] + bv;
            G[(size_t)mm * ncols + nn] = 1.0f / (1.0f + __expf(-lg));
        }
    }
}

// ---------- skinny GEMM (Bu): C[M x ncols] = A * B^T, one 16x16 tile / wave ----------
__global__ void k_gemm(const uint16_t* __restrict__ Abf,
                       const uint16_t* __restrict__ Bbf,
                       float* __restrict__ C, int K, int ncols) {
    int lane = threadIdx.x;
    int m0 = blockIdx.x * 16;
    int n0 = blockIdx.y * 16;
    int r    = lane & 15;
    int half = lane >> 4;
    const uint16_t* arow = Abf + (size_t)(m0 + r) * K;
    const uint16_t* brow = Bbf + (size_t)(n0 + r) * K;

    v8f acc = {};
    for (int k0 = 0; k0 < K; k0 += 32) {
        Frag a, b;
#pragma unroll
        for (int g = 0; g < 8; ++g) {
            int kk = k0 + ((g & 3) * 2) + ((g >> 2) * 16) + half * 8;
            a.u[g] = *(const uint32_t*)(arow + kk);
            b.u[g] = *(const uint32_t*)(brow + kk);
        }
        acc = wmma_bf16(a, b, acc);
    }
#pragma unroll
    for (int i = 0; i < 8; ++i) {
        int mm = m0 + i + half * 8;
        C[(size_t)mm * ncols + n0 + r] = acc[i];
    }
}

// ---------- causal decay scan: h[n] = Abar*h[n-1] + Bu[n]*Bbar ----------
__global__ void k_scan(const float* __restrict__ Bu,
                       const float* __restrict__ Abar, const float* __restrict__ Bbar,
                       float* __restrict__ H, int N) {
    int s = threadIdx.x;
    int b = blockIdx.x;
    if (s >= DS) return;
    float ab = Abar[s], bb = Bbar[s];
    float h = 0.0f;
    size_t base = (size_t)b * N * DS + s;
    for (int n = 0; n < N; ++n) {
        float bu = Bu[base + (size_t)n * DS] * bb;   // load independent of h -> pipelines
        h = fmaf(ab, h, bu);
        H[base + (size_t)n * DS] = fminf(fmaxf(h, -10.0f), 10.0f);
    }
}

// ---------- fused: y = h@W_C^T + D*x ; gate mix ; LayerNorm ----------
__global__ void k_fuse(const float* __restrict__ x, const float* __restrict__ H,
                       const float* __restrict__ W_C, const float* __restrict__ Dv,
                       const float* __restrict__ gate,
                       const float* __restrict__ gamma, const float* __restrict__ beta,
                       float* __restrict__ out) {
    int row = blockIdx.x;       // flattened (b, n)
    int m   = threadIdx.x;      // 0..511
    __shared__ float hrow[DS];
    __shared__ float red1[16], red2[16];

    if (m < DS) hrow[m] = H[(size_t)row * DS + m];
    __syncthreads();

    const float* wc = W_C + (size_t)m * DS;
    float acc = 0.0f;
#pragma unroll
    for (int s = 0; s < DS; ++s) acc = fmaf(hrow[s], wc[s], acc);

    float xv   = x[(size_t)row * DM + m];
    float ypre = acc + Dv[m] * xv;
    float g    = gate[(size_t)row * DM + m];
    float y    = g * ypre + (1.0f - g) * xv;

    // LayerNorm over 512: wave32 shuffle reduce, then cross-wave via LDS
    float s1 = y, s2 = y * y;
#pragma unroll
    for (int off = 16; off > 0; off >>= 1) {
        s1 += __shfl_down(s1, off);
        s2 += __shfl_down(s2, off);
    }
    int wave = m >> 5, lid = m & 31;
    if (lid == 0) { red1[wave] = s1; red2[wave] = s2; }
    __syncthreads();
    if (m < 32) {                       // whole wave 0 participates (valid shuffles)
        s1 = (m < 16) ? red1[m] : 0.0f;
        s2 = (m < 16) ? red2[m] : 0.0f;
#pragma unroll
        for (int off = 8; off > 0; off >>= 1) {
            s1 += __shfl_down(s1, off);
            s2 += __shfl_down(s2, off);
        }
        if (m == 0) { red1[0] = s1; red2[0] = s2; }
    }
    __syncthreads();

    float mean = red1[0] * (1.0f / DM);
    float var  = red2[0] * (1.0f / DM) - mean * mean;
    float inv  = rsqrtf(var + 1e-5f);
    out[(size_t)row * DM + m] = (y - mean) * inv * gamma[m] + beta[m];
}

extern "C" void kernel_launch(void* const* d_in, const int* in_sizes, int n_in,
                              void* d_out, int out_size, void* d_ws, size_t ws_size,
                              hipStream_t stream) {
    (void)n_in; (void)out_size; (void)ws_size;
    const float* x      = (const float*)d_in[0];
    const float* A_log  = (const float*)d_in[1];
    const float* W_B    = (const float*)d_in[2];
    const float* W_C    = (const float*)d_in[3];
    const float* Dv     = (const float*)d_in[4];
    const float* W_gate = (const float*)d_in[5];
    const float* b_gate = (const float*)d_in[6];
    const float* gamma  = (const float*)d_in[7];
    const float* beta   = (const float*)d_in[8];
    float*       out    = (float*)d_out;

    const int BN = in_sizes[0] / DM;      // B*N rows (8192)
    const int N  = 2048;                  // sequence length (reference-fixed)
    const int Bb = BN / N;                // batch (4)

    // ---- workspace layout ----
    float*    gateb  = (float*)d_ws;                          // BN*DM (gate values)
    float*    Bu     = gateb  + (size_t)BN * DM;              // BN*DS
    float*    Hbuf   = Bu     + (size_t)BN * DS;              // BN*DS
    float*    Abar   = Hbuf   + (size_t)BN * DS;              // 16
    float*    Bbar   = Abar + DS;                             // 16
    uint16_t* xh     = (uint16_t*)(Bbar + DS);                // BN*DM bf16
    uint16_t* Wgh    = xh  + (size_t)BN * DM;                 // DM*DM bf16
    uint16_t* WBh    = Wgh + (size_t)DM * DM;                 // DS*DM bf16

    k_setup<<<1, 32, 0, stream>>>(A_log, Abar, Bbar, 1.0f / (float)N);

    k_cvt<<<((size_t)BN * DM + 255) / 256, 256, 0, stream>>>(x, xh, BN * DM);
    k_cvt<<<(DM * DM + 255) / 256,        256, 0, stream>>>(W_gate, Wgh, DM * DM);
    k_cvt<<<(DS * DM + 255) / 256,        256, 0, stream>>>(W_B, WBh, DS * DM);

    // gate: (BN x DM) = sigmoid(xh * W_gate^T + b_gate) -- dominant GEMM, 2x2 WMMA/wave
    k_gemm_gate<<<dim3(BN / 64, DM / 128), 256, 0, stream>>>(xh, Wgh, b_gate, gateb, DM, DM);
    // Bu: (BN x DS) = xh * W_B^T
    k_gemm<<<dim3(BN / 16, DS / 16), 32, 0, stream>>>(xh, WBh, Bu, DM, DS);

    k_scan<<<Bb, 32, 0, stream>>>(Bu, Abar, Bbar, Hbuf, N);

    k_fuse<<<BN, DM, 0, stream>>>(x, Hbuf, W_C, Dv, gateb, gamma, beta, out);
}
